// GRUDecoder_42167988913005
// MI455X (gfx1250) — compile-verified, matrix operand order checked by reading, not run on hardware
//
#include <hip/hip_runtime.h>
#include <cstdint>
#include <cstddef>

typedef __bf16 bf16;
typedef bf16  v16bf __attribute__((ext_vector_type(16)));
typedef float v8f   __attribute__((ext_vector_type(8)));

#define B_   1024
#define T_   16
#define V_   8000
#define E_   256
#define H_   512
#define G3   1536
#define VPAD 8064   // 63 * 128, so logits B-tiles need no load guards

#define LDS_STRIDE 40  // 32 K-elems + 8 pad (80B rows) -> conflict-free ds_load_b128

union FragU { v16bf v; uint4 u[2]; };

// Async-copy one 128x32 bf16 tile (row-major, K contiguous) into LDS.
// 512 16B chunks; 256 threads issue 2 global_load_async_to_lds_b128 each.
__device__ __forceinline__ void issue_tile_async(const bf16* __restrict__ gsrc, int ld,
                                                 int row0, int k0, bf16* lbase) {
    const int tid = threadIdx.x;
#pragma unroll
    for (int it = 0; it < 2; ++it) {
        const int c    = tid + it * 256;
        const int row  = c >> 2;
        const int part = c & 3;
        const bf16* g  = gsrc + (size_t)(row0 + row) * (size_t)ld + k0 + part * 8;
        const uint32_t l = (uint32_t)(uintptr_t)(lbase + row * LDS_STRIDE + part * 8);
        asm volatile("global_load_async_to_lds_b128 %0, %1, off"
                     :: "v"(l), "v"((uint64_t)(uintptr_t)g) : "memory");
    }
}

// 16x32 bf16 fragment from an LDS tile (A-layout == B-layout on gfx1250):
// lanes 0-15 : row = row0+lane,    K = 0..7  and 16..23
// lanes 16-31: row = row0+lane-16, K = 8..15 and 24..31
__device__ __forceinline__ v16bf lds_frag(const bf16* lbase, int row0) {
    const int lane = threadIdx.x & 31;
    const bf16* p = lbase + (row0 + (lane & 15)) * LDS_STRIDE + ((lane >> 4) << 3);
    FragU f;
    f.u[0] = *reinterpret_cast<const uint4*>(p);
    f.u[1] = *reinterpret_cast<const uint4*>(p + 16);
    return f.v;
}

// C[M,N] = A[M,K] * B[N,K]^T + bias. A/B bf16 row-major (K contiguous), C fp32.
// Block = 256 threads = 8 waves; block tile 128x128; wave tile 32x64.
// K-tiles of 32 staged through double-buffered LDS with async copies (ASYNCcnt).
__global__ __launch_bounds__(256)
void wmma_gemm_bias(const bf16* __restrict__ A, int lda,
                    const bf16* __restrict__ Bm, int ldb,
                    const float* __restrict__ bias,
                    float* __restrict__ C, int ldc,
                    int K, int Nreal) {
    __shared__ bf16 Asm[2][128 * LDS_STRIDE];
    __shared__ bf16 Bsm[2][128 * LDS_STRIDE];

    const int lane = threadIdx.x & 31;
    const int wave = threadIdx.x >> 5;
    const int wm   = wave & 3;          // 4 waves along M
    const int wn   = wave >> 2;         // 2 waves along N
    const int rowBase = blockIdx.y * 128;
    const int colBase = blockIdx.x * 128;

    v8f acc[2][4];
#pragma unroll
    for (int i = 0; i < 2; ++i)
#pragma unroll
        for (int j = 0; j < 4; ++j) acc[i][j] = v8f{};

    const int ktiles = K >> 5;
    issue_tile_async(A,  lda, rowBase, 0, Asm[0]);
    issue_tile_async(Bm, ldb, colBase, 0, Bsm[0]);

    for (int kt = 0; kt < ktiles; ++kt) {
        const int cur = kt & 1;
        if (kt + 1 < ktiles) {
            issue_tile_async(A,  lda, rowBase, (kt + 1) << 5, Asm[cur ^ 1]);
            issue_tile_async(Bm, ldb, colBase, (kt + 1) << 5, Bsm[cur ^ 1]);
            // 4 ops just issued per wave; wait for the previous stage's 4.
            asm volatile("s_wait_asynccnt 4" ::: "memory");
        } else {
            asm volatile("s_wait_asynccnt 0" ::: "memory");
        }
        __syncthreads();   // whole tile visible to all waves

        v16bf a0 = lds_frag(Asm[cur], wm * 32);
        v16bf a1 = lds_frag(Asm[cur], wm * 32 + 16);
        v16bf b0 = lds_frag(Bsm[cur], wn * 64);
        v16bf b1 = lds_frag(Bsm[cur], wn * 64 + 16);
        v16bf b2 = lds_frag(Bsm[cur], wn * 64 + 32);
        v16bf b3 = lds_frag(Bsm[cur], wn * 64 + 48);

        acc[0][0] = __builtin_amdgcn_wmma_f32_16x16x32_bf16(false, a0, false, b0, (short)0, acc[0][0], false, false);
        acc[0][1] = __builtin_amdgcn_wmma_f32_16x16x32_bf16(false, a0, false, b1, (short)0, acc[0][1], false, false);
        acc[0][2] = __builtin_amdgcn_wmma_f32_16x16x32_bf16(false, a0, false, b2, (short)0, acc[0][2], false, false);
        acc[0][3] = __builtin_amdgcn_wmma_f32_16x16x32_bf16(false, a0, false, b3, (short)0, acc[0][3], false, false);
        acc[1][0] = __builtin_amdgcn_wmma_f32_16x16x32_bf16(false, a1, false, b0, (short)0, acc[1][0], false, false);
        acc[1][1] = __builtin_amdgcn_wmma_f32_16x16x32_bf16(false, a1, false, b1, (short)0, acc[1][1], false, false);
        acc[1][2] = __builtin_amdgcn_wmma_f32_16x16x32_bf16(false, a1, false, b2, (short)0, acc[1][2], false, false);
        acc[1][3] = __builtin_amdgcn_wmma_f32_16x16x32_bf16(false, a1, false, b3, (short)0, acc[1][3], false, false);

        __syncthreads();   // done reading buf[cur] before next stage overwrites it
    }

    // C layout: element e of lane l -> row = e + 8*(l>>4), col = l&15 (per 16x16 tile)
    const int hi = lane >> 4;
    const int cl = lane & 15;
#pragma unroll
    for (int i = 0; i < 2; ++i) {
#pragma unroll
        for (int j = 0; j < 4; ++j) {
            const int col = colBase + wn * 64 + j * 16 + cl;
            if (col < Nreal) {
                const float bv = bias ? bias[col] : 0.f;
                const int r0 = rowBase + wm * 32 + i * 16 + hi * 8;
#pragma unroll
                for (int e = 0; e < 8; ++e)
                    C[(size_t)(r0 + e) * (size_t)ldc + col] = acc[i][j][e] + bv;
            }
        }
    }
}

__global__ void f2bf(const float* __restrict__ s, bf16* __restrict__ d, int n) {
    int i = blockIdx.x * blockDim.x + threadIdx.x;
    if (i < n) d[i] = (bf16)s[i];
}

__global__ void f2bf_pad(const float* __restrict__ s, bf16* __restrict__ d,
                         int rows_real, int cols, int rows_pad) {
    int i = blockIdx.x * blockDim.x + threadIdx.x;
    if (i >= rows_pad * cols) return;
    int r = i / cols;
    d[i] = (r < rows_real) ? (bf16)s[i] : (bf16)0.f;
}

// x_t: t==0 -> embed[0] (SOS); t>0 -> embed[target[b][t-1]].  Rows ordered t*B+b.
__global__ void gather_x(const float* __restrict__ embed, const int* __restrict__ tgt,
                         bf16* __restrict__ Xb) {
    int idx = blockIdx.x * blockDim.x + threadIdx.x;
    if (idx >= T_ * B_ * E_) return;
    int e = idx % E_;
    int m = idx / E_;
    int b = m % B_;
    int t = m / B_;
    int token = (t == 0) ? 0 : tgt[b * T_ + (t - 1)];
    Xb[idx] = (bf16)embed[(size_t)token * E_ + e];
}

__global__ void init_h(const float* __restrict__ ctx, float* __restrict__ h,
                       bf16* __restrict__ hbf) {
    int i = blockIdx.x * blockDim.x + threadIdx.x;
    if (i < B_ * H_) { float v = ctx[i]; h[i] = v; hbf[i] = (bf16)v; }
}

// PyTorch GRU gate math [r,z,n]; updates h in place, emits bf16 h for next-step
// GEMM and into Hall at row b*T+t (matches d_out row order for the logits GEMM).
__global__ void gru_gates(const float* __restrict__ gi, const float* __restrict__ gh,
                          float* __restrict__ h, bf16* __restrict__ hbf,
                          bf16* __restrict__ Hall, int t) {
    int idx = blockIdx.x * blockDim.x + threadIdx.x;
    if (idx >= B_ * H_) return;
    int b = idx / H_;
    int j = idx % H_;
    const float* gib = gi + (size_t)b * G3;
    const float* ghb = gh + (size_t)b * G3;
    float ir = gib[j], iz = gib[H_ + j], in_ = gib[2 * H_ + j];
    float hr = ghb[j], hz = ghb[H_ + j], hn  = ghb[2 * H_ + j];
    float r  = 1.f / (1.f + __expf(-(ir + hr)));
    float z  = 1.f / (1.f + __expf(-(iz + hz)));
    float n  = tanhf(in_ + r * hn);
    float hp = (1.f - z) * n + z * h[idx];
    h[idx]   = hp;
    hbf[idx] = (bf16)hp;
    Hall[((size_t)b * T_ + t) * H_ + j] = (bf16)hp;
}

// In-place log-softmax per row; whole 8000-float row cached in LDS (32KB << 320KB/WGP)
// so d_out is read once and written once.
__global__ __launch_bounds__(256)
void log_softmax_rows(float* __restrict__ out) {
    __shared__ float srow[V_];
    __shared__ float red[256];
    const int row = blockIdx.x;
    float* base = out + (size_t)row * V_;
    const int tid = threadIdx.x;

    float lmax = -INFINITY;
    for (int i = tid; i < V_; i += 256) {
        float v = base[i]; srow[i] = v; lmax = fmaxf(lmax, v);
    }
    red[tid] = lmax; __syncthreads();
    for (int s = 128; s > 0; s >>= 1) {
        if (tid < s) red[tid] = fmaxf(red[tid], red[tid + s]);
        __syncthreads();
    }
    const float m = red[0];
    __syncthreads();

    float lsum = 0.f;
    for (int i = tid; i < V_; i += 256) lsum += __expf(srow[i] - m);
    red[tid] = lsum; __syncthreads();
    for (int s = 128; s > 0; s >>= 1) {
        if (tid < s) red[tid] += red[tid + s];
        __syncthreads();
    }
    const float ls = __logf(red[0]) + m;

    for (int i = tid; i < V_; i += 256) base[i] = srow[i] - ls;
}

extern "C" void kernel_launch(void* const* d_in, const int* in_sizes, int n_in,
                              void* d_out, int out_size, void* d_ws, size_t ws_size,
                              hipStream_t stream) {
    (void)in_sizes; (void)n_in; (void)out_size; (void)ws_size;
    const float* ctx   = (const float*)d_in[0];
    const int*   tgt   = (const int*)d_in[1];
    const float* embed = (const float*)d_in[2];
    const float* W_ih  = (const float*)d_in[3];
    const float* W_hh  = (const float*)d_in[4];
    const float* b_ih  = (const float*)d_in[5];
    const float* b_hh  = (const float*)d_in[6];
    const float* out_W = (const float*)d_in[7];
    const float* out_b = (const float*)d_in[8];
    float* out = (float*)d_out;

    char* ws = (char*)d_ws;
    size_t off = 0;
    auto alloc = [&](size_t bytes) -> char* {
        char* p = ws + off; off += (bytes + 255) & ~size_t(255); return p;
    };
    bf16*  Wihb = (bf16*)alloc((size_t)G3 * E_ * 2);
    bf16*  Whhb = (bf16*)alloc((size_t)G3 * H_ * 2);
    bf16*  oWb  = (bf16*)alloc((size_t)VPAD * H_ * 2);
    bf16*  Xb   = (bf16*)alloc((size_t)T_ * B_ * E_ * 2);
    bf16*  Hall = (bf16*)alloc((size_t)T_ * B_ * H_ * 2);
    float* hf   = (float*)alloc((size_t)B_ * H_ * 4);
    bf16*  hbf  = (bf16*)alloc((size_t)B_ * H_ * 2);
    float* gi   = (float*)alloc((size_t)B_ * G3 * 4);
    float* gh   = (float*)alloc((size_t)B_ * G3 * 4);

    dim3 blk(256);
    f2bf<<<(G3 * E_ + 255) / 256, blk, 0, stream>>>(W_ih, Wihb, G3 * E_);
    f2bf<<<(G3 * H_ + 255) / 256, blk, 0, stream>>>(W_hh, Whhb, G3 * H_);
    f2bf_pad<<<(VPAD * H_ + 255) / 256, blk, 0, stream>>>(out_W, oWb, V_, H_, VPAD);
    gather_x<<<(T_ * B_ * E_ + 255) / 256, blk, 0, stream>>>(embed, tgt, Xb);
    init_h<<<(B_ * H_ + 255) / 256, blk, 0, stream>>>(ctx, hf, hbf);

    for (int t = 0; t < T_; ++t) {
        // gi = X_t @ W_ih^T + b_ih   (M=1024, N=1536, K=256)
        wmma_gemm_bias<<<dim3(G3 / 128, B_ / 128), blk, 0, stream>>>(
            Xb + (size_t)t * B_ * E_, E_, Wihb, E_, b_ih, gi, G3, E_, G3);
        // gh = h   @ W_hh^T + b_hh   (M=1024, N=1536, K=512)
        wmma_gemm_bias<<<dim3(G3 / 128, B_ / 128), blk, 0, stream>>>(
            hbf, H_, Whhb, H_, b_hh, gh, G3, H_, G3);
        gru_gates<<<(B_ * H_ + 255) / 256, blk, 0, stream>>>(gi, gh, hf, hbf, Hall, t);
    }

    // logits = Hall @ out_W^T + out_b  (M=16384, Npad=8064, K=512) -> d_out rows (b*T+t)
    wmma_gemm_bias<<<dim3(VPAD / 128, (T_ * B_) / 128), blk, 0, stream>>>(
        Hall, H_, oWb, H_, out_b, out, V_, H_, V_);

    log_softmax_rows<<<T_ * B_, blk, 0, stream>>>(out);
}
